// LSTMCellModel_23648089931864
// MI455X (gfx1250) — compile-verified, matrix operand order checked by reading, not run on hardware
//
#include <hip/hip_runtime.h>
#include <hip/hip_bf16.h>

// ---------------------------------------------------------------------------
// Types for CDNA5 WMMA bf16 fragments (wave32)
// ---------------------------------------------------------------------------
typedef __bf16 bf16;
typedef __attribute__((ext_vector_type(16))) __bf16 v16bf;
typedef __attribute__((ext_vector_type(8)))  __bf16 v8bf;
typedef __attribute__((ext_vector_type(8)))  float  v8f;

#define INPUT_DIM  512
#define HIDDEN_DIM 1024
#define GATES_DIM  4096   // 4*HIDDEN_DIM
#define SEQ        1024
#define STEPS      128
#define OUT_DIM    1000

// ---------------------------------------------------------------------------
// Fragment loaders.
// bf16 16x32 A/B fragment layout (ISA 7.12.2):
//   lane l: row (A) / col (B) = base + (l & 15)
//   elements j=0..7  -> K = k0 + 8*(l>=16) + j
//   elements j=8..15 -> K = k0 + 16 + 8*(l>=16) + (j-8)
// Both chunks are 16B-contiguous -> two global_load_b128 per fragment.
// ---------------------------------------------------------------------------
__device__ __forceinline__ v16bf frag_combine(v8bf lo, v8bf hi) {
    v16bf r;
#pragma unroll
    for (int i = 0; i < 8; ++i) { r[i] = lo[i]; r[8 + i] = hi[i]; }
    return r;
}

__device__ __forceinline__ v16bf load_frag_bf16(const bf16* __restrict__ base,
                                                int ldk, int row, int k0, int lane) {
    const int half = (lane >> 4) & 1;
    const bf16* p = base + (size_t)row * (size_t)ldk + k0 + half * 8;
    v8bf lo = *reinterpret_cast<const v8bf*>(p);
    v8bf hi = *reinterpret_cast<const v8bf*>(p + 16);
    return frag_combine(lo, hi);
}

// A fragment sourced from f32 (x slice), converted + scaled on the fly.
__device__ __forceinline__ v16bf load_frag_f32(const float* __restrict__ base,
                                               int ldk, int row, int k0, int lane,
                                               float scale) {
    const int half = (lane >> 4) & 1;
    const float* p = base + (size_t)row * (size_t)ldk + k0 + half * 8;
    v8bf lo, hi;
#pragma unroll
    for (int i = 0; i < 8; ++i) {
        lo[i] = (bf16)(p[i] * scale);
        hi[i] = (bf16)(p[i + 16] * scale);
    }
    return frag_combine(lo, hi);
}

__device__ __forceinline__ float sigmoid_f(float x) {
    return 1.0f / (1.0f + __expf(-x));
}
__device__ __forceinline__ float tanh_f(float x) {
    x = fminf(fmaxf(x, -15.0f), 15.0f);
    float e = __expf(2.0f * x);
    return (e - 1.0f) / (e + 1.0f);
}

// ---------------------------------------------------------------------------
// Fused LSTM cell: gates GEMM (bf16 WMMA, f32 accum) + pointwise cell update.
// Grid: (SEQ/256, HIDDEN_DIM/16), block: 256 threads (8 waves).
// Wave w owns 32 rows (two 16-row A tiles) at m0 = bx*256 + w*32; all waves
// share cols n0 = by*16 and compute that 16-col slice for all four gates
// (i,f,g,o at n, n+1H, n+2H, n+3H), so:
//   - each B fragment is reused for 2 WMMAs (two row tiles),
//   - each A fragment is reused for 4 WMMAs (four gates),
//   - inner loop: 6 fragment loads per 8 WMMAs,
//   - the nonlinearity is fused with zero gates-matrix traffic.
// ---------------------------------------------------------------------------
__global__ __launch_bounds__(256)
void lstm_cell_wmma(const float* __restrict__ xt,      // [SEQ, INPUT_DIM] f32
                    const bf16*  __restrict__ w_ih,    // [GATES_DIM, INPUT_DIM]
                    const bf16*  __restrict__ w_hh,    // [GATES_DIM, HIDDEN_DIM]
                    const float* __restrict__ bsum,    // [GATES_DIM] (b_ih+b_hh)
                    const bf16*  __restrict__ h_in,    // [SEQ, HIDDEN_DIM]
                    bf16*        __restrict__ h_out,   // [SEQ, HIDDEN_DIM]
                    float*       __restrict__ c,       // [SEQ, HIDDEN_DIM] in/out
                    bf16*        __restrict__ hlast)   // [HIDDEN_DIM] or null
{
    const int lane = threadIdx.x & 31;
    const int wave = threadIdx.x >> 5;
    const int m0   = blockIdx.x * 256 + wave * 32;     // two 16-row tiles
    const int n0   = blockIdx.y * 16;
    const int arow = m0 + (lane & 15);
    const int bcol = n0 + (lane & 15);

    v8f acc[2][4] = {};   // [row tile][gate]

    // Phase 1: x * W_ih^T, K = INPUT_DIM (x scaled by 255 during bf16 convert)
    for (int k0 = 0; k0 < INPUT_DIM; k0 += 32) {
        v16bf a0 = load_frag_f32(xt, INPUT_DIM, arow,      k0, lane, 255.0f);
        v16bf a1 = load_frag_f32(xt, INPUT_DIM, arow + 16, k0, lane, 255.0f);
#pragma unroll
        for (int g = 0; g < 4; ++g) {
            v16bf b = load_frag_bf16(w_ih, INPUT_DIM, g * HIDDEN_DIM + bcol, k0, lane);
            acc[0][g] = __builtin_amdgcn_wmma_f32_16x16x32_bf16(
                false, a0, false, b, (short)0, acc[0][g], false, false);
            acc[1][g] = __builtin_amdgcn_wmma_f32_16x16x32_bf16(
                false, a1, false, b, (short)0, acc[1][g], false, false);
        }
    }
    // Phase 2: h * W_hh^T, K = HIDDEN_DIM
    for (int k0 = 0; k0 < HIDDEN_DIM; k0 += 32) {
        v16bf a0 = load_frag_bf16(h_in, HIDDEN_DIM, arow,      k0, lane);
        v16bf a1 = load_frag_bf16(h_in, HIDDEN_DIM, arow + 16, k0, lane);
#pragma unroll
        for (int g = 0; g < 4; ++g) {
            v16bf b = load_frag_bf16(w_hh, HIDDEN_DIM, g * HIDDEN_DIM + bcol, k0, lane);
            acc[0][g] = __builtin_amdgcn_wmma_f32_16x16x32_bf16(
                false, a0, false, b, (short)0, acc[0][g], false, false);
            acc[1][g] = __builtin_amdgcn_wmma_f32_16x16x32_bf16(
                false, a1, false, b, (short)0, acc[1][g], false, false);
        }
    }

    // Pointwise cell update. C/D layout: m = tile_m0 + r + 8*(lane>=16),
    // n = n0 + (lane&15).
    const int half = lane >> 4;
    const int n    = n0 + (lane & 15);
    const float bi  = bsum[0 * HIDDEN_DIM + n];
    const float bff = bsum[1 * HIDDEN_DIM + n];
    const float bg  = bsum[2 * HIDDEN_DIM + n];
    const float bo  = bsum[3 * HIDDEN_DIM + n];

#pragma unroll
    for (int tt = 0; tt < 2; ++tt) {
        const int tm0 = m0 + 16 * tt;
#pragma unroll
        for (int r = 0; r < 8; ++r) {
            const int m = tm0 + r + 8 * half;
            const float iv = sigmoid_f(acc[tt][0][r] + bi);
            const float fv = sigmoid_f(acc[tt][1][r] + bff);
            const float gv = tanh_f  (acc[tt][2][r] + bg);
            const float ov = sigmoid_f(acc[tt][3][r] + bo);
            const size_t idx = (size_t)m * HIDDEN_DIM + n;
            const float cn = fv * c[idx] + iv * gv;
            c[idx] = cn;
            const float hn = ov * tanh_f(cn);
            const bf16 hb = (bf16)hn;
            h_out[idx] = hb;
            if (hlast != nullptr && m == SEQ - 1) hlast[n] = hb;
        }
    }
}

// ---------------------------------------------------------------------------
// Final FC: out[128,1000] = hlast[128,1024] @ w_fc^T + b_fc  (WMMA)
// Grid (128/16, ceil(1000/16)=63), block 32 (one wave -> one 16x16 tile).
// ---------------------------------------------------------------------------
__global__ __launch_bounds__(32)
void fc_wmma(const bf16* __restrict__ hlast,  // [STEPS, HIDDEN_DIM]
             const bf16* __restrict__ wfc,    // [OUT_DIM, HIDDEN_DIM]
             const float* __restrict__ bfc,   // [OUT_DIM]
             float* __restrict__ out)         // [STEPS, OUT_DIM]
{
    const int lane = threadIdx.x & 31;
    const int m0 = blockIdx.x * 16;
    const int n0 = blockIdx.y * 16;
    const int arow = m0 + (lane & 15);
    int brow = n0 + (lane & 15);
    if (brow > OUT_DIM - 1) brow = OUT_DIM - 1;   // clamp OOB B rows (stores guarded)

    v8f acc = {};
    for (int k0 = 0; k0 < HIDDEN_DIM; k0 += 32) {
        v16bf a = load_frag_bf16(hlast, HIDDEN_DIM, arow, k0, lane);
        v16bf b = load_frag_bf16(wfc,   HIDDEN_DIM, brow, k0, lane);
        acc = __builtin_amdgcn_wmma_f32_16x16x32_bf16(
            false, a, false, b, (short)0, acc, false, false);
    }

    const int half = lane >> 4;
    const int n = n0 + (lane & 15);
    if (n < OUT_DIM) {
        const float bv = bfc[n];
#pragma unroll
        for (int r = 0; r < 8; ++r) {
            const int m = m0 + r + 8 * half;
            out[(size_t)m * OUT_DIM + n] = acc[r] + bv;
        }
    }
}

// ---------------------------------------------------------------------------
// Prep kernels: f32->bf16 weight conversion, bias fold, state zero-init.
// ---------------------------------------------------------------------------
__global__ void cvt_f32_to_bf16(bf16* __restrict__ dst, const float* __restrict__ src, int n) {
    int i = blockIdx.x * blockDim.x + threadIdx.x;
    if (i < n) dst[i] = (bf16)src[i];
}
__global__ void bias_sum(float* __restrict__ dst, const float* __restrict__ a,
                         const float* __restrict__ b, int n) {
    int i = blockIdx.x * blockDim.x + threadIdx.x;
    if (i < n) dst[i] = a[i] + b[i];
}
__global__ void init_state(bf16* __restrict__ h, float* __restrict__ c, int n) {
    int i = blockIdx.x * blockDim.x + threadIdx.x;
    if (i < n) { h[i] = (bf16)0.0f; c[i] = 0.0f; }
}

// ---------------------------------------------------------------------------
// Launcher
// ---------------------------------------------------------------------------
extern "C" void kernel_launch(void* const* d_in, const int* in_sizes, int n_in,
                              void* d_out, int out_size, void* d_ws, size_t ws_size,
                              hipStream_t stream) {
    const float* x     = (const float*)d_in[0];
    const float* w_ih1 = (const float*)d_in[1];
    const float* w_hh1 = (const float*)d_in[2];
    const float* b_ih1 = (const float*)d_in[3];
    const float* b_hh1 = (const float*)d_in[4];
    const float* w_ih2 = (const float*)d_in[5];
    const float* w_hh2 = (const float*)d_in[6];
    const float* b_ih2 = (const float*)d_in[7];
    const float* b_hh2 = (const float*)d_in[8];
    const float* w_fc  = (const float*)d_in[9];
    const float* b_fc  = (const float*)d_in[10];
    float* out = (float*)d_out;

    // Workspace carve-up (~36 MB), 256B aligned slices.
    char* ws = (char*)d_ws;
    size_t off = 0;
    auto take = [&](size_t bytes) -> char* {
        char* p = ws + off;
        off = (off + bytes + 255) & ~(size_t)255;
        return p;
    };
    const int NIH = GATES_DIM * INPUT_DIM;    // 2,097,152
    const int NHH = GATES_DIM * HIDDEN_DIM;   // 4,194,304
    const int NFC = OUT_DIM * HIDDEN_DIM;     // 1,024,000
    const int NST = SEQ * HIDDEN_DIM;         // 1,048,576

    bf16* w_ih1b = (bf16*)take((size_t)NIH * 2);
    bf16* w_hh1b = (bf16*)take((size_t)NHH * 2);
    bf16* w_ih2b = (bf16*)take((size_t)NIH * 2);
    bf16* w_hh2b = (bf16*)take((size_t)NHH * 2);
    bf16* w_fcb  = (bf16*)take((size_t)NFC * 2);
    float* bsum1 = (float*)take((size_t)GATES_DIM * 4);
    float* bsum2 = (float*)take((size_t)GATES_DIM * 4);
    bf16* hA     = (bf16*)take((size_t)NST * 2);
    bf16* hB     = (bf16*)take((size_t)NST * 2);
    float* cbuf  = (float*)take((size_t)NST * 4);
    bf16* hlast  = (bf16*)take((size_t)STEPS * HIDDEN_DIM * 2);

    const int T = 256;
    cvt_f32_to_bf16<<<(NIH + T - 1) / T, T, 0, stream>>>(w_ih1b, w_ih1, NIH);
    cvt_f32_to_bf16<<<(NHH + T - 1) / T, T, 0, stream>>>(w_hh1b, w_hh1, NHH);
    cvt_f32_to_bf16<<<(NIH + T - 1) / T, T, 0, stream>>>(w_ih2b, w_ih2, NIH);
    cvt_f32_to_bf16<<<(NHH + T - 1) / T, T, 0, stream>>>(w_hh2b, w_hh2, NHH);
    cvt_f32_to_bf16<<<(NFC + T - 1) / T, T, 0, stream>>>(w_fcb, w_fc, NFC);
    bias_sum<<<(GATES_DIM + T - 1) / T, T, 0, stream>>>(bsum1, b_ih1, b_hh1, GATES_DIM);
    bias_sum<<<(GATES_DIM + T - 1) / T, T, 0, stream>>>(bsum2, b_ih2, b_hh2, GATES_DIM);
    init_state<<<(NST + T - 1) / T, T, 0, stream>>>(hA, cbuf, NST);

    // Sequential scan: both cells per step; h ping-pongs, c in-place.
    bf16* hbuf[2] = {hA, hB};
    const dim3 grid(SEQ / 256, HIDDEN_DIM / 16);
    for (int t = 0; t < STEPS; ++t) {
        const float* xt = x + (size_t)t * SEQ * INPUT_DIM;
        const int q = 2 * t;
        // layer 1: reads h[q&1], writes h[(q+1)&1]
        lstm_cell_wmma<<<grid, 256, 0, stream>>>(
            xt, w_ih1b, w_hh1b, bsum1,
            hbuf[q & 1], hbuf[(q + 1) & 1], cbuf, nullptr);
        // layer 2: input is xt again (per reference), state is layer-1 output
        lstm_cell_wmma<<<grid, 256, 0, stream>>>(
            xt, w_ih2b, w_hh2b, bsum2,
            hbuf[(q + 1) & 1], hbuf[q & 1], cbuf,
            hlast + (size_t)t * HIDDEN_DIM);
    }

    fc_wmma<<<dim3(STEPS / 16, (OUT_DIM + 15) / 16), 32, 0, stream>>>(
        hlast, w_fcb, b_fc, out);
}
